// GAT_DTI_1511828488569
// MI455X (gfx1250) — matrix operand order repaired and checked by reading, not compiled, not run on hardware
//
#include <hip/hip_runtime.h>
#include <hip/hip_bf16.h>
#include <math.h>

// Problem constants (from reference)
#define NODES  50000
#define EDGES  800000
#define NGRAPH 256
#define ETOT   (EDGES + NODES)   // edges + self loops

typedef __attribute__((ext_vector_type(16))) _Float16     v16h;
typedef __attribute__((ext_vector_type(8)))  _Float16     v8h;
typedef __attribute__((ext_vector_type(8)))  float        v8f;
typedef __attribute__((ext_vector_type(4)))  unsigned int v4u;
typedef __attribute__((ext_vector_type(4)))  int          v4i;
typedef __attribute__((ext_vector_type(8)))  int          v8i;

// ---------------------------------------------------------------------------
// float atomic max via int/uint punning (works with -inf init)
__device__ __forceinline__ void atomicMaxF(float* addr, float val) {
  if (val >= 0.0f) atomicMax((int*)addr, __float_as_int(val));
  else             atomicMin((unsigned int*)addr, __float_as_uint(val));
}

__device__ __forceinline__ int src_of(const int* __restrict__ ei, int e) {
  return (e < EDGES) ? ei[e] : (e - EDGES);
}
__device__ __forceinline__ void edge_sd(const int* __restrict__ ei, int e, int& s, int& d) {
  if (e < EDGES) { s = ei[e]; d = ei[EDGES + e]; }
  else           { s = e - EDGES; d = s; }          // self loop
}

// ===========================================================================
// CSR-by-dst construction (built once; identical for all three layers)
// ===========================================================================
__global__ void zero_int_k(int* __restrict__ p, int n) {
  int idx = blockIdx.x * blockDim.x + threadIdx.x;
  if (idx < n) p[idx] = 0;
}

__global__ void deg_k(const int* __restrict__ ei, int* __restrict__ deg) {
  int e = blockIdx.x * blockDim.x + threadIdx.x;
  if (e >= ETOT) return;
  int s, d; edge_sd(ei, e, s, d);
  atomicAdd(&deg[d], 1);
}

// Single-block exclusive scan over NODES entries (Hillis-Steele per 1024 chunk)
__global__ __launch_bounds__(1024)
void scan_k(const int* __restrict__ deg, int* __restrict__ rowptr) {
  __shared__ int sh[1024];
  __shared__ int carry;
  int tid = threadIdx.x;
  if (tid == 0) carry = 0;
  __syncthreads();
  for (int base = 0; base < NODES; base += 1024) {
    int i = base + tid;
    int v = (i < NODES) ? deg[i] : 0;
    sh[tid] = v;
    __syncthreads();
    for (int s = 1; s < 1024; s <<= 1) {
      int t = (tid >= s) ? sh[tid - s] : 0;
      __syncthreads();
      sh[tid] += t;
      __syncthreads();
    }
    if (i < NODES) rowptr[i] = carry + sh[tid] - v;   // exclusive
    __syncthreads();
    if (tid == 1023) carry += sh[1023];
    __syncthreads();
  }
  if (tid == 0) rowptr[NODES] = carry;                // == ETOT
}

__global__ void copy_int_k(const int* __restrict__ a, int* __restrict__ b, int n) {
  int idx = blockIdx.x * blockDim.x + threadIdx.x;
  if (idx < n) b[idx] = a[idx];
}

__global__ void fill_k(const int* __restrict__ ei, int* __restrict__ cursor,
                       int* __restrict__ csr) {
  int e = blockIdx.x * blockDim.x + threadIdx.x;
  if (e >= ETOT) return;
  int s, d; edge_sd(ei, e, s, d);
  int pos = atomicAdd(&cursor[d], 1);
  csr[pos] = e;
}

// Insertion-sort each node's edge list -> deterministic accumulation order
__global__ void sort_k(const int* __restrict__ rowptr, int* __restrict__ csr) {
  int n = blockIdx.x * blockDim.x + threadIdx.x;
  if (n >= NODES) return;
  int st = rowptr[n], en = rowptr[n + 1];
  for (int i = st + 1; i < en; ++i) {
    int key = csr[i];
    int j = i - 1;
    while (j >= st && csr[j] > key) { csr[j + 1] = csr[j]; --j; }
    csr[j + 1] = key;
  }
}

// ===========================================================================
// GEMM: prep + WMMA with TDM-staged B panel in LDS
// ===========================================================================
// Transpose + f32->f16 convert weights: w [K,M] row-major -> wt [M,K] f16
__global__ void prep_w_k(const float* __restrict__ w, _Float16* __restrict__ wt,
                         int K, int M) {
  int idx = blockIdx.x * blockDim.x + threadIdx.x;
  if (idx >= K * M) return;
  int k = idx / M, m = idx % M;
  wt[(size_t)m * K + k] = (_Float16)w[(size_t)k * M + m];
}

// f32 -> f16 convert activations
__global__ void prep_xh_k(const float* __restrict__ X, _Float16* __restrict__ xh,
                          size_t n) {
  size_t idx = (size_t)blockIdx.x * blockDim.x + threadIdx.x;
  if (idx < n) xh[idx] = (_Float16)X[idx];
}

// C[nrows,M] f32 = XH[nrows,K] f16  @  WT[M,K]^T f16
// Block: 256 thr = 8 waves; wave w does rows [row0,row0+16) x 64 cols.
// B panel (64 x K halves, <=32KB) staged in LDS once per block via TDM.
__global__ __launch_bounds__(256)
void gat_gemm_wmma(const _Float16* __restrict__ XH, const _Float16* __restrict__ WT,
                   float* __restrict__ C, int nrows, int K, int M) {
  __shared__ _Float16 shB[64 * 256];
  const int lane = threadIdx.x & 31;
  const int wave = threadIdx.x >> 5;
  const int g    = lane >> 4;
  const int lm   = lane & 15;
  const int row0 = blockIdx.x * 128 + wave * 16;
  const int col0 = blockIdx.y * 64;

#if __has_builtin(__builtin_amdgcn_tensor_load_to_lds) && __has_builtin(__builtin_amdgcn_s_wait_tensorcnt)
  if (threadIdx.x == 0) {
    // ISA 10.2: low 32 bits of a flat LDS-aperture address == LDS byte offset
    unsigned lds_off = (unsigned)(uintptr_t)(void*)shB;
    // D# group0: count=1 | lds_addr | global_addr | type=2
    unsigned long long ga = (unsigned long long)(WT + (size_t)col0 * K);
    v4u g0;
    g0[0] = 1u;                                   // count=1, user descriptor
    g0[1] = lds_off;                              // lds_addr (bytes)
    g0[2] = (unsigned)ga;                         // global_addr[31:0]
    g0[3] = (unsigned)(ga >> 32) | 0x80000000u;   // global_addr[56:32] | type=2
    // D# group1: 2D tile K x 64, data_size=2B, stride=K elements
    v8i g1;
    g1[0] = 0x00010000;                 // workgroup_mask=0, data_size=1 (2B)
    g1[1] = (int)((unsigned)K << 16);   // tensor_dim0[15:0]=K
    g1[2] = (int)(64u << 16);           // tensor_dim0 hi=0, tensor_dim1[15:0]=64
    g1[3] = (int)((unsigned)K << 16);   // tensor_dim1 hi=0, tile_dim0=K
    g1[4] = 64;                         // tile_dim1=64, tile_dim2=0
    g1[5] = K;                          // tensor_dim0_stride[31:0]=K
    g1[6] = 0;                          // stride hi, tensor_dim1_stride lo
    g1[7] = 0;
    v4i z4 = {0, 0, 0, 0};
#if defined(__clang_major__) && (__clang_major__ >= 23)
    v8i z8 = {0, 0, 0, 0, 0, 0, 0, 0};
    __builtin_amdgcn_tensor_load_to_lds(g0, g1, z4, z4, z8, 0);
#else
    __builtin_amdgcn_tensor_load_to_lds(g0, g1, z4, z4, 0);
#endif
    __builtin_amdgcn_s_wait_tensorcnt(0);
  }
  __syncthreads();
  // The TDM builtin carries no pointer argument, so the compiler cannot see
  // that shB was written. Escape the pointer + full memory clobber so the
  // ds_loads of the B fragments below are not folded away.
  {
    void* p = (void*)shB;
    asm volatile("" : "+v"(p) : : "memory");
  }
#else
  // Fallback: cooperative copy of the B panel into LDS
  {
    const v8h* src = (const v8h*)(WT + (size_t)col0 * K);
    v8h* dst = (v8h*)shB;
    int total = 64 * K / 8;
    for (int i = threadIdx.x; i < total; i += 256) dst[i] = src[i];
  }
  __syncthreads();
#endif

  int m = row0 + lm;
  if (m > nrows - 1) m = nrows - 1;            // clamp reads; stores guarded
  const _Float16* xrow = XH + (size_t)m * K;

  v8f acc[4] = {};
  for (int k0 = 0; k0 < K; k0 += 32) {
    // A fragment: halves e -> K=k0+g*16+e
    v16h a = *(const v16h*)(xrow + k0 + g * 16);
#pragma unroll
    for (int t = 0; t < 4; ++t) {
      // B fragment from LDS: lane holds col n=t*16+lm, halves K=k0+g*16..+15
      const _Float16* bp = shB + (size_t)(t * 16 + lm) * K + k0 + g * 16;
      v16h b = *(const v16h*)bp;
      acc[t] = __builtin_amdgcn_wmma_f32_16x16x32_f16(
          false, a, false, b, (short)0, acc[t], false, false);
    }
  }
  // C/D layout: VGPR r -> row row0 + g*8 + r, col = col0 + t*16 + lm
#pragma unroll
  for (int t = 0; t < 4; ++t) {
    int col = col0 + t * 16 + lm;
#pragma unroll
    for (int r = 0; r < 8; ++r) {
      int row = row0 + g * 8 + r;
      if (row < nrows) C[(size_t)row * M + col] = acc[t][r];
    }
  }
}

// ===========================================================================
// Attention: logits, per-node softmax stats, per-edge alpha, CSR aggregation
// ===========================================================================
__global__ void logits_k(const float* __restrict__ Hf, const float* __restrict__ as,
                         const float* __restrict__ ad, float* __restrict__ als,
                         float* __restrict__ ald, int H, int M) {
  int idx = blockIdx.x * blockDim.x + threadIdx.x;
  if (idx >= NODES * H) return;
  int n = idx / H, h = idx % H;
  const float* row = Hf + (size_t)n * M + h * 64;
  const float* a1 = as + h * 64;
  const float* a2 = ad + h * 64;
  float s1 = 0.f, s2 = 0.f;
#pragma unroll 8
  for (int c = 0; c < 64; ++c) { float v = row[c]; s1 += v * a1[c]; s2 += v * a2[c]; }
  als[idx] = s1; ald[idx] = s2;
}

// Per-(node,head) softmax max & denominator over the node's CSR edge list
__global__ void node_softmax_k(const int* __restrict__ ei, const int* __restrict__ rowptr,
                               const int* __restrict__ csr, const float* __restrict__ als,
                               const float* __restrict__ ald, float* __restrict__ emaxA,
                               float* __restrict__ denA, int H) {
  int idx = blockIdx.x * blockDim.x + threadIdx.x;
  if (idx >= NODES * H) return;
  int n = idx / H, h = idx % H;
  int st = rowptr[n], en = rowptr[n + 1];
  float dl = ald[idx];
  float mx = -INFINITY;
  for (int i = st; i < en; ++i) {
    int s = src_of(ei, csr[i]);
    float v = als[s * H + h] + dl;
    v = v < 0.f ? 0.2f * v : v;
    mx = fmaxf(mx, v);
  }
  if (!(mx > -1e38f)) mx = 0.f;      // isfinite guard (matches reference)
  float den = 0.f;
  for (int i = st; i < en; ++i) {
    int s = src_of(ei, csr[i]);
    float v = als[s * H + h] + dl;
    v = v < 0.f ? 0.2f * v : v;
    den += __expf(v - mx);
  }
  emaxA[idx] = mx;
  denA[idx]  = den;
}

// alpha[e,h] = exp(lrelu - emax[dst]) / (den[dst] + eps)
__global__ void alpha_k(const int* __restrict__ ei, const float* __restrict__ als,
                        const float* __restrict__ ald, const float* __restrict__ emaxA,
                        const float* __restrict__ denA, float* __restrict__ alpha, int H) {
  int idx = blockIdx.x * blockDim.x + threadIdx.x;
  if (idx >= ETOT * H) return;
  int e = idx / H, h = idx % H;
  int s, d; edge_sd(ei, e, s, d);
  float v = als[s * H + h] + ald[d * H + h];
  v = v < 0.f ? 0.2f * v : v;
  alpha[idx] = __expf(v - emaxA[d * H + h]) / (denA[d * H + h] + 1e-16f);
}

// out[n,:] = sum_{e in CSR[n]} alpha[e,h] * h[src(e),:]   (no atomics)
__global__ void agg_k(const int* __restrict__ ei, const int* __restrict__ rowptr,
                      const int* __restrict__ csr, const float* __restrict__ Hf,
                      const float* __restrict__ alpha, float* __restrict__ out,
                      int H, int M) {
  int n = blockIdx.x;
  int c = threadIdx.x;               // 0..M-1
  int h = c >> 6;                    // 64 channels per head
  int st = rowptr[n], en = rowptr[n + 1];
  float acc = 0.f;
  for (int i = st; i < en; ++i) {
    int e = csr[i];
    int s = src_of(ei, e);
    acc += alpha[e * H + h] * Hf[(size_t)s * M + c];
  }
  out[(size_t)n * M + c] = acc;
}

// bias + batchnorm + ELU, in place
__global__ void post_k(float* __restrict__ X, const float* __restrict__ b,
                       const float* __restrict__ g, const float* __restrict__ be,
                       const float* __restrict__ mean, const float* __restrict__ var,
                       int M) {
  size_t idx = (size_t)blockIdx.x * blockDim.x + threadIdx.x;
  if (idx >= (size_t)NODES * M) return;
  int m = (int)(idx % M);
  float v = X[idx] + b[m];
  v = (v - mean[m]) * rsqrtf(var[m] + 1e-5f) * g[m] + be[m];
  X[idx] = v > 0.f ? v : expm1f(v);
}

// ===========================================================================
// Readout
// ===========================================================================
__global__ void pool_init_k(float* __restrict__ gsum, float* __restrict__ gmax,
                            float* __restrict__ gcnt) {
  int idx = blockIdx.x * blockDim.x + threadIdx.x;
  if (idx < NGRAPH * 64) { gsum[idx] = 0.f; gmax[idx] = -INFINITY; }
  if (idx < NGRAPH) gcnt[idx] = 0.f;
}

__global__ void pool_k(const float* __restrict__ X, const int* __restrict__ batch,
                       float* __restrict__ gsum, float* __restrict__ gmax,
                       float* __restrict__ gcnt) {
  int idx = blockIdx.x * blockDim.x + threadIdx.x;
  if (idx >= NODES * 64) return;
  int n = idx >> 6, c = idx & 63;
  int g = batch[n];
  float v = X[(size_t)n * 64 + c];
  atomicAdd(&gsum[g * 64 + c], v);
  atomicMaxF(&gmax[g * 64 + c], v);
  if (c == 0) atomicAdd(&gcnt[g], 1.0f);
}

// One 64-thread block per graph: [mean|max|sum] (192) -> 64 -> 32 -> 1
__global__ __launch_bounds__(64)
void head_k(const float* __restrict__ gsum, const float* __restrict__ gmax,
            const float* __restrict__ gcnt,
            const float* __restrict__ fc1w, const float* __restrict__ fc1b,
            const float* __restrict__ fc2w, const float* __restrict__ fc2b,
            const float* __restrict__ fc3w, const float* __restrict__ fc3b,
            float* __restrict__ out) {
  __shared__ float feat[192];
  __shared__ float h1[64];
  __shared__ float h2[32];
  int g = blockIdx.x, c = threadIdx.x;
  float cnt = fmaxf(gcnt[g], 1.0f);
  float s  = gsum[g * 64 + c];
  float mx = gmax[g * 64 + c];
  if (!(mx > -1e38f)) mx = 0.f;
  feat[c]       = s / cnt;   // mean
  feat[64 + c]  = mx;        // max
  feat[128 + c] = s;         // sum
  __syncthreads();
  float acc = fc1b[c];
  for (int i = 0; i < 192; ++i) acc += feat[i] * fc1w[i * 64 + c];
  h1[c] = fmaxf(acc, 0.f);
  __syncthreads();
  if (c < 32) {
    float a2 = fc2b[c];
    for (int i = 0; i < 64; ++i) a2 += h1[i] * fc2w[i * 32 + c];
    h2[c] = fmaxf(a2, 0.f);
  }
  __syncthreads();
  if (c == 0) {
    float a3 = fc3b[0];
    for (int i = 0; i < 32; ++i) a3 += h2[i] * fc3w[i];
    out[g] = a3;
  }
}

// ===========================================================================
extern "C" void kernel_launch(void* const* d_in, const int* in_sizes, int n_in,
                              void* d_out, int out_size, void* d_ws, size_t ws_size,
                              hipStream_t stream) {
  (void)in_sizes; (void)n_in; (void)out_size; (void)ws_size;
  const float* x     = (const float*)d_in[0];
  const int*   ei    = (const int*)d_in[1];
  const int*   batch = (const int*)d_in[2];
  const float* W[3]  = {(const float*)d_in[3],  (const float*)d_in[11], (const float*)d_in[19]};
  const float* AS[3] = {(const float*)d_in[4],  (const float*)d_in[12], (const float*)d_in[20]};
  const float* AD[3] = {(const float*)d_in[5],  (const float*)d_in[13], (const float*)d_in[21]};
  const float* Bb[3] = {(const float*)d_in[6],  (const float*)d_in[14], (const float*)d_in[22]};
  const float* Gg[3] = {(const float*)d_in[7],  (const float*)d_in[15], (const float*)d_in[23]};
  const float* BE[3] = {(const float*)d_in[8],  (const float*)d_in[16], (const float*)d_in[24]};
  const float* Mn[3] = {(const float*)d_in[9],  (const float*)d_in[17], (const float*)d_in[25]};
  const float* Vr[3] = {(const float*)d_in[10], (const float*)d_in[18], (const float*)d_in[26]};
  const float* fc1w = (const float*)d_in[27];
  const float* fc1b = (const float*)d_in[28];
  const float* fc2w = (const float*)d_in[29];
  const float* fc2b = (const float*)d_in[30];
  const float* fc3w = (const float*)d_in[31];
  const float* fc3b = (const float*)d_in[32];

  char* ws = (char*)d_ws;
  size_t off = 0;
  auto carve = [&](size_t bytes) -> char* {
    char* p = ws + off;
    off += (bytes + 255) & ~(size_t)255;
    return p;
  };
  float*    buf0   = (float*)carve((size_t)NODES * 256 * 4);  // X / aggregated out
  float*    buf1   = (float*)carve((size_t)NODES * 256 * 4);  // h = x@w
  _Float16* xh     = (_Float16*)carve((size_t)NODES * 256 * 2);
  float*    alpha  = (float*)carve((size_t)ETOT * 4 * 4);
  _Float16* wt     = (_Float16*)carve(256 * 256 * 2);
  float*    als    = (float*)carve((size_t)NODES * 4 * 4);
  float*    ald    = (float*)carve((size_t)NODES * 4 * 4);
  float*    emaxA  = (float*)carve((size_t)NODES * 4 * 4);
  float*    denA   = (float*)carve((size_t)NODES * 4 * 4);
  int*      deg    = (int*)carve((size_t)NODES * 4);
  int*      rowptr = (int*)carve((size_t)(NODES + 1) * 4);
  int*      cursor = (int*)carve((size_t)NODES * 4);
  int*      csr    = (int*)carve((size_t)ETOT * 4);
  float*    gsum   = (float*)carve(NGRAPH * 64 * 4);
  float*    gmax   = (float*)carve(NGRAPH * 64 * 4);
  float*    gcnt   = (float*)carve(NGRAPH * 4);

  // ---- CSR by dst (once, shared by all layers) ----
  zero_int_k<<<(NODES + 255) / 256, 256, 0, stream>>>(deg, NODES);
  deg_k<<<(ETOT + 255) / 256, 256, 0, stream>>>(ei, deg);
  scan_k<<<1, 1024, 0, stream>>>(deg, rowptr);
  copy_int_k<<<(NODES + 255) / 256, 256, 0, stream>>>(rowptr, cursor, NODES);
  fill_k<<<(ETOT + 255) / 256, 256, 0, stream>>>(ei, cursor, csr);
  sort_k<<<(NODES + 255) / 256, 256, 0, stream>>>(rowptr, csr);

  // ---- layers ----
  const float* Xcur = x;
  for (int L = 0; L < 3; ++L) {
    const int K = (L == 0) ? 128 : 256;
    const int H = (L == 2) ? 1 : 4;
    const int M = H * 64;

    prep_w_k<<<(K * M + 255) / 256, 256, 0, stream>>>(W[L], wt, K, M);
    prep_xh_k<<<(unsigned)(((size_t)NODES * K + 255) / 256), 256, 0, stream>>>(
        Xcur, xh, (size_t)NODES * K);

    dim3 gg((NODES + 127) / 128, M / 64);
    gat_gemm_wmma<<<gg, 256, 0, stream>>>(xh, wt, buf1, NODES, K, M);

    logits_k<<<(NODES * H + 255) / 256, 256, 0, stream>>>(buf1, AS[L], AD[L], als, ald, H, M);
    node_softmax_k<<<(NODES * H + 255) / 256, 256, 0, stream>>>(
        ei, rowptr, csr, als, ald, emaxA, denA, H);
    alpha_k<<<(ETOT * H + 255) / 256, 256, 0, stream>>>(ei, als, ald, emaxA, denA, alpha, H);
    agg_k<<<NODES, M, 0, stream>>>(ei, rowptr, csr, buf1, alpha, buf0, H, M);

    post_k<<<(unsigned)(((size_t)NODES * M + 255) / 256), 256, 0, stream>>>(
        buf0, Bb[L], Gg[L], BE[L], Mn[L], Vr[L], M);
    Xcur = buf0;
  }

  // ---- readout ----
  pool_init_k<<<(NGRAPH * 64 + 255) / 256, 256, 0, stream>>>(gsum, gmax, gcnt);
  pool_k<<<(NODES * 64 + 255) / 256, 256, 0, stream>>>(buf0, batch, gsum, gmax, gcnt);
  head_k<<<NGRAPH, 64, 0, stream>>>(gsum, gmax, gcnt, fc1w, fc1b, fc2w, fc2b,
                                    fc3w, fc3b, (float*)d_out);
}
